// DecoderMultiHeadAttention_37245956391252
// MI455X (gfx1250) — compile-verified, matrix operand order checked by reading, not run on hardware
//
#include <hip/hip_runtime.h>

#define BB 2
#define TT 2048
#define CC 768
#define HH 12
#define DD 64
#define BT (BB * TT)  // 4096

typedef __attribute__((ext_vector_type(16))) _Float16 v16h;
typedef __attribute__((ext_vector_type(8)))  _Float16 v8h;
typedef __attribute__((ext_vector_type(4)))  _Float16 v4h;
typedef __attribute__((ext_vector_type(2)))  _Float16 h2;
typedef __attribute__((ext_vector_type(8)))  float    v8f;
typedef __attribute__((ext_vector_type(4)))  float    v4f;

static __device__ __forceinline__ v16h cat16(v8h lo, v8h hi) {
  v16h r;
#pragma unroll
  for (int i = 0; i < 8; ++i) { r[i] = lo[i]; r[i + 8] = hi[i]; }
  return r;
}

// ---------------------------------------------------------------------------
// f32 -> f16 conversion (vectorized x4; all our sizes are multiples of 4)
// ---------------------------------------------------------------------------
__global__ void cvt_f32_to_f16_kernel(const float* __restrict__ in,
                                      _Float16* __restrict__ out, int n) {
  int i = (blockIdx.x * blockDim.x + threadIdx.x) * 4;
  if (i < n) {
    v4f v = *(const v4f*)(in + i);
    v4h o;
#pragma unroll
    for (int j = 0; j < 4; ++j) o[j] = (_Float16)v[j];
    *(v4h*)(out + i) = o;
  }
}

// ---------------------------------------------------------------------------
// Tiled WMMA GEMM: C[M][N] = A[M][K] * Bw[K][ldb] (with output-column remap).
// Block tile 128x128, 8 waves (4x2), wave tile 32x64, K-step 32.
// outHalf=1 -> store f16 (no bias); outHalf=0 -> store f32 + bias.
// ---------------------------------------------------------------------------
__global__ __launch_bounds__(256) void gemm_wmma_kernel(
    const _Float16* __restrict__ A, const _Float16* __restrict__ Bw,
    void* __restrict__ Cout, const float* __restrict__ bias,
    int N, int K, int ldb, int remapAt, int remapAdd, int outHalf) {
  __shared__ __align__(16) _Float16 Al[128 * 32];   // [m][k]
  __shared__ __align__(16) _Float16 Btl[128 * 32];  // [n][k] (transposed)

  const int tid = threadIdx.x;
  const int lane = tid & 31, l16 = lane & 15, hi = (lane >> 4) & 1;
  const int wid = tid >> 5;
  const int mW = (wid & 3) * 32, nW = (wid >> 2) * 64;
  const int mB = blockIdx.y * 128;
  const int nB = blockIdx.x * 128;
  const int nSrc = nB + ((nB >= remapAt) ? remapAdd : 0);

  v8f c[2][4];
#pragma unroll
  for (int mi = 0; mi < 2; ++mi)
#pragma unroll
    for (int ni = 0; ni < 4; ++ni) c[mi][ni] = (v8f)0.0f;

  const int am = tid >> 1;            // 0..127 : row in A tile
  const int ak = (tid & 1) * 16;      // 0/16   : k offset in A tile
  const int bk = tid >> 4;            // 0..15  : k row in B tile
  const int bn = (tid & 15) * 8;      // 0..120 : n offset in B tile

  for (int k0 = 0; k0 < K; k0 += 32) {
    // --- cooperative loads into LDS ---
    {
      const _Float16* ga = A + (size_t)(mB + am) * K + k0 + ak;
      v8h x0 = *(const v8h*)(ga);
      v8h x1 = *(const v8h*)(ga + 8);
      *(v8h*)(&Al[am * 32 + ak])     = x0;
      *(v8h*)(&Al[am * 32 + ak + 8]) = x1;
      if (k0 + 32 < K) __builtin_prefetch((const void*)(ga + 32), 0, 0);
    }
#pragma unroll
    for (int kh = 0; kh < 2; ++kh) {
      const int kk = bk + kh * 16;
      const _Float16* gb = Bw + (size_t)(k0 + kk) * ldb + nSrc + bn;
      v8h x = *(const v8h*)(gb);
      if (k0 + 32 < K) __builtin_prefetch((const void*)(gb + (size_t)32 * ldb), 0, 0);
#pragma unroll
      for (int i = 0; i < 8; ++i) Btl[(bn + i) * 32 + kk] = x[i];
    }
    __syncthreads();

    // --- fragments + WMMA ---
    v16h a[2], b[4];
#pragma unroll
    for (int mi = 0; mi < 2; ++mi) {
      const int r = mW + mi * 16 + l16;
      a[mi] = cat16(*(const v8h*)(&Al[r * 32 + hi * 8]),
                    *(const v8h*)(&Al[r * 32 + 16 + hi * 8]));
    }
#pragma unroll
    for (int ni = 0; ni < 4; ++ni) {
      const int n = nW + ni * 16 + l16;
      b[ni] = cat16(*(const v8h*)(&Btl[n * 32 + hi * 8]),
                    *(const v8h*)(&Btl[n * 32 + 16 + hi * 8]));
    }
#pragma unroll
    for (int mi = 0; mi < 2; ++mi)
#pragma unroll
      for (int ni = 0; ni < 4; ++ni)
        c[mi][ni] = __builtin_amdgcn_wmma_f32_16x16x32_f16(
            false, a[mi], false, b[ni], (short)0, c[mi][ni], false, false);
    __syncthreads();
  }

  // --- epilogue ---
#pragma unroll
  for (int mi = 0; mi < 2; ++mi) {
#pragma unroll
    for (int ni = 0; ni < 4; ++ni) {
      const int col = nB + nW + ni * 16 + l16;
      const float bv = bias ? bias[col] : 0.0f;
#pragma unroll
      for (int j = 0; j < 8; ++j) {
        const int row = mB + mW + mi * 16 + j + 8 * hi;
        const float v = c[mi][ni][j];
        if (outHalf) ((_Float16*)Cout)[(size_t)row * N + col] = (_Float16)v;
        else         ((float*)Cout)[(size_t)row * N + col] = v + bv;
      }
    }
  }
}

// ---------------------------------------------------------------------------
// RoPE + head scatter. kv: [BT][2C] f16 (k-chunk | v-chunk).
// K = rope(k); Q = rope(v) * D^-1/2 (source bug is faithful; scale folded);
// V = v. Outputs in [B][H][T][D] f16.
// ---------------------------------------------------------------------------
__global__ void rope_scatter_kernel(const _Float16* __restrict__ kv,
                                    const float* __restrict__ cosT,
                                    const float* __restrict__ sinT,
                                    _Float16* __restrict__ Kh,
                                    _Float16* __restrict__ Qh,
                                    _Float16* __restrict__ Vh) {
  const int NP = DD / 2;  // 32 rotation pairs per head
  int idx = blockIdx.x * blockDim.x + threadIdx.x;
  int i = idx % NP;
  int h = (idx / NP) % HH;
  int t = (idx / (NP * HH)) % TT;
  int b = idx / (NP * HH * TT);
  if (b >= BB) return;
  const int row = b * TT + t;
  const float c = cosT[t * NP + i];
  const float s = sinT[t * NP + i];
  const int kcol = h * DD + 2 * i;
  const h2 kp = *(const h2*)(kv + (size_t)row * (2 * CC) + kcol);
  const h2 vp = *(const h2*)(kv + (size_t)row * (2 * CC) + CC + kcol);
  const float kr = (float)kp[0], ki = (float)kp[1];
  const float vr = (float)vp[0], vi = (float)vp[1];
  const float sc = 0.125f;  // 1/sqrt(64)
  const size_t obase = (((size_t)(b * HH + h)) * TT + t) * DD + 2 * i;
  h2 ko, qo, vo;
  ko[0] = (_Float16)(kr * c - ki * s); ko[1] = (_Float16)(kr * s + ki * c);
  qo[0] = (_Float16)((vr * c - vi * s) * sc); qo[1] = (_Float16)((vr * s + vi * c) * sc);
  vo[0] = (_Float16)vr;               vo[1] = (_Float16)vi;
  *(h2*)(Kh + obase) = ko;
  *(h2*)(Qh + obase) = qo;
  *(h2*)(Vh + obase) = vo;
}

// ---------------------------------------------------------------------------
// Flash attention. Block = (64 q rows, one (b,h)); 4 waves, 16 q rows each.
// Key tiles of 32; online softmax; everything on v_wmma_f32_16x16x32_f16.
// ---------------------------------------------------------------------------
__global__ __launch_bounds__(128) void attn_kernel(
    const _Float16* __restrict__ Qh, const _Float16* __restrict__ Kh,
    const _Float16* __restrict__ Vh, _Float16* __restrict__ Oh /* [BT][C] */) {
  __shared__ __align__(16) _Float16 Kl[32 * DD];      // [key][d]
  __shared__ __align__(16) _Float16 Vt[DD * 32];      // [d][key] (transposed)
  __shared__ __align__(16) _Float16 Pl[4][16 * 32];   // per-wave P scratch

  const int bh = blockIdx.y;
  const int b = bh / HH, h = bh % HH;
  const int qTile = blockIdx.x;
  const int tid = threadIdx.x;
  const int wid = tid >> 5, lane = tid & 31, l16 = lane & 15, hi = (lane >> 4) & 1;
  const _Float16* Qb = Qh + (size_t)bh * TT * DD;
  const _Float16* Kb = Kh + (size_t)bh * TT * DD;
  const _Float16* Vb = Vh + (size_t)bh * TT * DD;
  const int qBase = qTile * 64 + wid * 16;

  // Q fragments: rows qBase..qBase+15, D split into two K=32 slices
  const _Float16* qrow = Qb + (size_t)(qBase + l16) * DD;
  const v16h aQ0 = cat16(*(const v8h*)(qrow + hi * 8),      *(const v8h*)(qrow + 16 + hi * 8));
  const v16h aQ1 = cat16(*(const v8h*)(qrow + 32 + hi * 8), *(const v8h*)(qrow + 48 + hi * 8));

  float m[8], l[8];
  v8f o[4];
#pragma unroll
  for (int j = 0; j < 8; ++j) { m[j] = -1e30f; l[j] = 0.0f; }
#pragma unroll
  for (int n = 0; n < 4; ++n) o[n] = (v8f)0.0f;

  const int tok = tid & 31, d0 = (tid >> 5) * 16;  // cooperative-load mapping
  const int ktEnd = qTile * 2 + 2;
  const float LOG2E = 1.4426950408889634f;

  for (int kt = 0; kt < ktEnd; ++kt) {
    const int keyBase = kt * 32;
    {  // stage K (row-major) and V (transposed) tiles into LDS
      const _Float16* gk = Kb + (size_t)(keyBase + tok) * DD + d0;
      v8h k0 = *(const v8h*)(gk);
      v8h k1 = *(const v8h*)(gk + 8);
      *(v8h*)(&Kl[tok * DD + d0])     = k0;
      *(v8h*)(&Kl[tok * DD + d0 + 8]) = k1;
      const _Float16* gv = Vb + (size_t)(keyBase + tok) * DD + d0;
      v8h v0 = *(const v8h*)(gv);
      v8h v1 = *(const v8h*)(gv + 8);
#pragma unroll
      for (int i = 0; i < 8; ++i) {
        Vt[(d0 + i) * 32 + tok]     = v0[i];
        Vt[(d0 + 8 + i) * 32 + tok] = v1[i];
      }
    }
    __syncthreads();

    if (keyBase <= qBase + 15) {  // wave-uniform: EXEC stays all-ones for WMMA
      v8f sf[2];
#pragma unroll
      for (int kf = 0; kf < 2; ++kf) {
        const _Float16* krw = &Kl[(kf * 16 + l16) * DD];
        const v16h b0 = cat16(*(const v8h*)(krw + hi * 8),      *(const v8h*)(krw + 16 + hi * 8));
        const v16h b1 = cat16(*(const v8h*)(krw + 32 + hi * 8), *(const v8h*)(krw + 48 + hi * 8));
        v8f s = (v8f)0.0f;
        s = __builtin_amdgcn_wmma_f32_16x16x32_f16(false, aQ0, false, b0, (short)0, s, false, false);
        s = __builtin_amdgcn_wmma_f32_16x16x32_f16(false, aQ1, false, b1, (short)0, s, false, false);
        const int key = keyBase + kf * 16 + l16;
#pragma unroll
        for (int j = 0; j < 8; ++j) {
          const int row = qBase + j + 8 * hi;
          if (key > row) s[j] = -3.0e38f;   // causal mask (finite, NaN-safe)
        }
        sf[kf] = s;
      }
      float mnew[8], scale[8], pv[2][8];
#pragma unroll
      for (int j = 0; j < 8; ++j) {  // row max: xor-shuffles stay in 16-lane half
        float t = fmaxf(sf[0][j], sf[1][j]);
#pragma unroll
        for (int off = 8; off >= 1; off >>= 1) t = fmaxf(t, __shfl_xor(t, off, 32));
        mnew[j] = fmaxf(m[j], t);
        scale[j] = exp2f((m[j] - mnew[j]) * LOG2E);
      }
#pragma unroll
      for (int kf = 0; kf < 2; ++kf)
#pragma unroll
        for (int j = 0; j < 8; ++j)
          pv[kf][j] = exp2f((sf[kf][j] - mnew[j]) * LOG2E);
#pragma unroll
      for (int j = 0; j < 8; ++j) {
        float sum = pv[0][j] + pv[1][j];
#pragma unroll
        for (int off = 8; off >= 1; off >>= 1) sum += __shfl_xor(sum, off, 32);
        l[j] = l[j] * scale[j] + sum;
        m[j] = mnew[j];
      }
#pragma unroll
      for (int n = 0; n < 4; ++n)
#pragma unroll
        for (int j = 0; j < 8; ++j) o[n][j] *= scale[j];

      // P: C-layout -> A-layout via per-wave LDS round trip
      _Float16* P = &Pl[wid][0];
#pragma unroll
      for (int kf = 0; kf < 2; ++kf)
#pragma unroll
        for (int j = 0; j < 8; ++j)
          P[(j + 8 * hi) * 32 + kf * 16 + l16] = (_Float16)pv[kf][j];
      const v16h aP = cat16(*(const v8h*)(&P[l16 * 32 + hi * 8]),
                            *(const v8h*)(&P[l16 * 32 + 16 + hi * 8]));
#pragma unroll
      for (int n = 0; n < 4; ++n) {
        const int d = n * 16 + l16;
        const v16h bV = cat16(*(const v8h*)(&Vt[d * 32 + hi * 8]),
                              *(const v8h*)(&Vt[d * 32 + 16 + hi * 8]));
        o[n] = __builtin_amdgcn_wmma_f32_16x16x32_f16(false, aP, false, bV, (short)0, o[n], false, false);
      }
    }
    __syncthreads();
  }

  // normalize and store f16 to [B*T][C] (col = h*D + d) for the proj GEMM
#pragma unroll
  for (int n = 0; n < 4; ++n) {
    const int d = n * 16 + l16;
#pragma unroll
    for (int j = 0; j < 8; ++j) {
      const int tq = qBase + j + 8 * hi;
      Oh[(size_t)(b * TT + tq) * CC + h * DD + d] = (_Float16)(o[n][j] / l[j]);
    }
  }
}

// ---------------------------------------------------------------------------
// Driver
// ---------------------------------------------------------------------------
extern "C" void kernel_launch(void* const* d_in, const int* in_sizes, int n_in,
                              void* d_out, int out_size, void* d_ws, size_t ws_size,
                              hipStream_t stream) {
  (void)in_sizes; (void)n_in; (void)out_size; (void)ws_size;
  const float* x     = (const float*)d_in[0];
  const float* cosT  = (const float*)d_in[1];
  const float* sinT  = (const float*)d_in[2];
  const float* Watt  = (const float*)d_in[3];
  const float* Wproj = (const float*)d_in[4];
  const float* bproj = (const float*)d_in[5];
  float* out = (float*)d_out;

  char* ws = (char*)d_ws;
  size_t off = 0;
  auto take = [&](size_t elems) -> _Float16* {
    _Float16* p = (_Float16*)(ws + off);
    off += ((elems * sizeof(_Float16)) + 255) & ~(size_t)255;
    return p;
  };
  _Float16* Xh  = take((size_t)BT * CC);            // x in f16
  _Float16* Wah = take((size_t)CC * 3 * CC);        // W_att in f16
  _Float16* Wph = take((size_t)CC * CC);            // W_proj in f16
  _Float16* kvH = take((size_t)BT * 2 * CC);        // [BT][k|v] f16
  _Float16* Qd  = take((size_t)BB * HH * TT * DD);  // rope(v)*scale
  _Float16* Kd  = take((size_t)BB * HH * TT * DD);  // rope(k)
  _Float16* Vd  = take((size_t)BB * HH * TT * DD);  // v
  _Float16* Ah  = take((size_t)BT * CC);            // attention output f16

  const int thr = 256;
  {
    int n = BT * CC;
    cvt_f32_to_f16_kernel<<<dim3((n / 4 + thr - 1) / thr), thr, 0, stream>>>(x, Xh, n);
    n = CC * 3 * CC;
    cvt_f32_to_f16_kernel<<<dim3((n / 4 + thr - 1) / thr), thr, 0, stream>>>(Watt, Wah, n);
    n = CC * CC;
    cvt_f32_to_f16_kernel<<<dim3((n / 4 + thr - 1) / thr), thr, 0, stream>>>(Wproj, Wph, n);
  }
  // GEMM1: only k and v chunks of W_att (output col n>=C maps to source col n+C)
  gemm_wmma_kernel<<<dim3((2 * CC) / 128, BT / 128), 256, 0, stream>>>(
      Xh, Wah, (void*)kvH, nullptr, 2 * CC, CC, 3 * CC, CC, CC, 1);
  // RoPE + head scatter
  {
    int total = BB * TT * HH * (DD / 2);
    rope_scatter_kernel<<<dim3(total / thr), thr, 0, stream>>>(kvH, cosT, sinT, Kd, Qd, Vd);
  }
  // Flash attention
  attn_kernel<<<dim3(TT / 64, BB * HH), 128, 0, stream>>>(Qd, Kd, Vd, Ah);
  // Output projection + bias (f32 out)
  gemm_wmma_kernel<<<dim3(CC / 128, BT / 128), 256, 0, stream>>>(
      Ah, Wph, (void*)out, bproj, CC, CC, CC, 1 << 30, 0, 0);
}
